// type3_nufft_naive_81415400063467
// MI455X (gfx1250) — matrix operand order repaired and checked by reading, not compiled, not run on hardware
//
#include <hip/hip_runtime.h>
#include <hip/hip_bf16.h>

typedef __attribute__((ext_vector_type(16))) _Float16 v16h;
typedef __attribute__((ext_vector_type(8)))  float    v8f;

#define NUFFT_M   16384      // flattened image size (128*128)
#define NUFFT_K   8192       // trajectory length
#define NWAVES    8          // waves per block
#define MITERS    (NUFFT_M / (NWAVES * 32))   // 64 chunk steps per wave

__global__ __launch_bounds__(256)
void type3_nufft_wmma_kernel(const float* __restrict__ phis,    // (3, M)
                             const float* __restrict__ alphas,  // (3, K)
                             const float* __restrict__ xre,     // (8, M)
                             const float* __restrict__ xim,     // (8, M)
                             float* __restrict__ out)           // (8, K) complex interleaved
{
    __shared__ float red[NWAVES][32][16];   // 16 KB partial accumulators

    const int lane = threadIdx.x & 31;
    const int wave = threadIdx.x >> 5;
    const int col  = lane & 15;             // output column / B column within tile
    const bool hi  = lane >= 16;
    const int kg   = blockIdx.x * 16 + col; // global k index

    // Per-lane trajectory coefficients, pre-scaled by -2*pi.
    const float TWO_PI = 6.28318530717958647692f;
    const float w0 = -TWO_PI * alphas[0 * NUFFT_K + kg];
    const float w1 = -TWO_PI * alphas[1 * NUFFT_K + kg];
    const float w2 = -TWO_PI * alphas[2 * NUFFT_K + kg];

    // A-matrix row for this lane: rows 0..7 = x_re, rows 8..15 = x_im (ISA A layout: M = lane%16)
    const float* xsrc = (col < 8) ? (xre + col * NUFFT_M) : (xim + (col - 8) * NUFFT_M);
    const int aoff = hi ? 16 : 0;   // A: lanes 0-15 hold K 0..15; lanes 16-31 K 16..31
    const int boff = hi ? 16 : 0;   // B: lanes 0-15 hold K rows 0..15; lanes 16-31 rows 16..31

    v8f accC = {};   // [Xre;Xim] @ cos
    v8f accS = {};   // [Xre;Xim] @ sin

    const int mstart = wave * MITERS * 32;

    for (int it = 0; it < MITERS; ++it) {
        const int mbase = mstart + it * 32;

        // Prefetch next chunk of phis into cache hierarchy.
        if (it + 1 < MITERS) {
            __builtin_prefetch(phis + mbase + 32 + boff, 0, 3);
            __builtin_prefetch(phis + NUFFT_M + mbase + 32 + boff, 0, 3);
            __builtin_prefetch(phis + 2 * NUFFT_M + mbase + 32 + boff, 0, 3);
        }

        // ---- Build A fragment (16x32 f16): this lane's row, 16 k-slots ----
        v16h a;
#pragma unroll
        for (int i = 0; i < 8; ++i) {
            a[i]     = (_Float16)xsrc[mbase + aoff + i];
            a[8 + i] = (_Float16)xsrc[mbase + 8 + aoff + i];
        }

        // ---- Build B fragments (32x16 f16): cos & sin of the phase column ----
        const float* p0 = phis + 0 * NUFFT_M + mbase + boff;
        const float* p1 = phis + 1 * NUFFT_M + mbase + boff;
        const float* p2 = phis + 2 * NUFFT_M + mbase + boff;
        v16h bc, bs;
#pragma unroll
        for (int i = 0; i < 16; ++i) {
            float ang = fmaf(w0, p0[i], fmaf(w1, p1[i], w2 * p2[i]));
            bc[i] = (_Float16)__cosf(ang);
            bs[i] = (_Float16)__sinf(ang);
        }

        // ---- Two WMMAs per chunk: accumulate real-basis products ----
        accC = __builtin_amdgcn_wmma_f32_16x16x32_f16(
                   false, a, false, bc, (short)0, accC, false, false);
        accS = __builtin_amdgcn_wmma_f32_16x16x32_f16(
                   false, a, false, bs, (short)0, accS, false, false);
    }

    // ---- Cross-wave reduction through LDS ----
#pragma unroll
    for (int j = 0; j < 8; ++j) {
        red[wave][lane][j]     = accC[j];
        red[wave][lane][8 + j] = accS[j];
    }
    __syncthreads();

    if (wave == 0) {
        float sc[8], ss[8];
#pragma unroll
        for (int j = 0; j < 8; ++j) {
            float c = 0.f, s = 0.f;
#pragma unroll
            for (int w = 0; w < NWAVES; ++w) {
                c += red[w][lane][j];
                s += red[w][lane][8 + j];
            }
            sc[j] = c; ss[j] = s;
        }

        // Accumulator layout: lanes 0-15 hold rows 0..7 (Xre@*), lanes 16-31 rows 8..15 (Xim@*).
        // Y_re = Xre@C - Xim@S ; Y_im = Xre@S + Xim@C  -> exchange S-partials across halves.
#pragma unroll
        for (int j = 0; j < 8; ++j) {
            float sOther = __shfl_xor(ss[j], 16, 32);
            float val = hi ? (sOther + sc[j])    // lanes 16-31: imag of row j
                           : (sc[j] - sOther);   // lanes  0-15: real of row j
            out[(size_t)(j * NUFFT_K + kg) * 2 + (hi ? 1 : 0)] = val;
        }
    }
}

extern "C" void kernel_launch(void* const* d_in, const int* in_sizes, int n_in,
                              void* d_out, int out_size, void* d_ws, size_t ws_size,
                              hipStream_t stream) {
    (void)in_sizes; (void)n_in; (void)d_ws; (void)ws_size; (void)out_size;
    const float* phis   = (const float*)d_in[0];   // (3,128,128)
    const float* alphas = (const float*)d_in[1];   // (3,8192)
    const float* xre    = (const float*)d_in[2];   // (8,128,128)
    const float* xim    = (const float*)d_in[3];   // (8,128,128)
    float* out          = (float*)d_out;           // (8,8192) complex64 interleaved

    dim3 grid(NUFFT_K / 16);   // 512 k-tiles
    dim3 block(32 * NWAVES);   // 8 waves
    type3_nufft_wmma_kernel<<<grid, block, 0, stream>>>(phis, alphas, xre, xim, out);
}
